// MonolithicDecoderLayer_44461501448411
// MI455X (gfx1250) — matrix operand order repaired and checked by reading, not compiled, not run on hardware
//
#include <hip/hip_runtime.h>
#include <math.h>

// ---------------- model constants ----------------
#define TSEQ    1024
#define HID     2048
#define HEADS   16
#define Q_LORA  1536
#define KV_LORA 512
#define NOPE    128
#define ROPE_D  64
#define VDIM    128
#define QKD     (NOPE + ROPE_D)   // 192
#define N_EXP   8
#define MOE_I   1408
#define SH_I    2816
#define QKV_N   (Q_LORA + KV_LORA + ROPE_D) // 2112
#define EPSV    1e-6f
#define ROPE_BASE 10000.0f

typedef __bf16 bf16;
typedef __attribute__((ext_vector_type(16))) __bf16 v16bf;
typedef __attribute__((ext_vector_type(8)))  float  v8f;

// ---- native bf16 conversion (v_cvt_pk_bf16_f32 on CDNA5) ----
__device__ __forceinline__ bf16 f2bf(float x) { return (bf16)x; }

__device__ __forceinline__ unsigned pk2bf(float x, float y) {
#if __has_builtin(__builtin_amdgcn_cvt_pk_bf16_f32)
    auto r = __builtin_amdgcn_cvt_pk_bf16_f32(x, y);
    unsigned u; __builtin_memcpy(&u, &r, sizeof(unsigned));
    return u;
#else
    union { bf16 b[2]; unsigned u; } c;
    c.b[0] = (bf16)x; c.b[1] = (bf16)y;
    return c.u;
#endif
}

// WMMA fragment k order used here (wave32, v_wmma_f32_16x16x32_bf16; not AMD's table order, A and B match):
//  A (16x32): lane holds row m = lane&15; half = lane>>4;
//             element e -> K = half*16 + e
//             => one contiguous 16-elem K-run
//  B (32x16): lane holds col n = lane&15; element e -> K = half*16 + e
//             => one contiguous 16-elem K-run
//  C/D f32  : lane holds col n = lane&15; vgpr i -> row m = i + 8*half

#define LDS_STRIDE 40   // bf16 elements per LDS row: 80B, 16B-aligned, bank-spread

__device__ __forceinline__ v16bf ld_frag_a(const bf16* As, int row, int half) {
    union { uint4 q[2]; v16bf v; } u;
    u.q[0] = *(const uint4*)(As + row * LDS_STRIDE + half * 16);
    u.q[1] = *(const uint4*)(As + row * LDS_STRIDE + half * 16 + 8);
    return u.v;
}
__device__ __forceinline__ v16bf ld_frag_b(const bf16* Bs, int col, int half) {
    union { uint4 q[2]; v16bf v; } u;
    const uint4* p = (const uint4*)(Bs + col * LDS_STRIDE + half * 16);
    u.q[0] = p[0]; u.q[1] = p[1];
    return u.v;
}

// ---------------- epilogue modes (compile-time) ----------------
#define EP_STORE    0  // D = acc
#define EP_ADD      1  // D = acc + aux[idx]          (residual add)
#define EP_SILUMUL  2  // D = silu(aux[idx]) * acc    (gated MLP)
#define EP_SCALEACC 3  // D[idx] += scalev[row*ss]*acc (MoE combine)

// =======================================================================
// Double-buffered, software-pipelined WMMA GEMM:
//   D[M,N] = A[M,K] x B[K,N], fp32 in/out, bf16 matrix math.
// Block = 256 threads (8 waves) computing a 128 x WN tile; wave w owns
// rows w*16..w*16+15 across all WN columns (WN/16 accumulators).
// Per 32-deep K step: coalesced float2 global loads (prefetched one step
// ahead, overlapping WMMA) -> v_cvt_pk_bf16_f32 -> LDS (A row-major,
// B transposed with k-pairs packed into ds_store_b32) -> ds_load_b128
// fragments -> WMMA.  One barrier per K step (double buffer).
// Requires M%128==0, N%WN==0, K%32==0.
// =======================================================================
template <int WN, int MODE>
__global__ __launch_bounds__(256)
void wmma_gemm(const float* __restrict__ A, int lda,
               const float* __restrict__ B, int ldb,
               float* __restrict__ D, int ldd,
               int K,
               const float* __restrict__ aux,
               const float* __restrict__ scalev, int sstride)
{
    constexpr int NT  = WN / 16;  // accumulators per wave
    constexpr int NCB = WN / 16;  // B columns staged per thread (8 or 4)

    const int tid  = threadIdx.x;
    const int lane = tid & 31;
    const int wave = tid >> 5;
    const int half = lane >> 4;
    const int l15  = lane & 15;
    const int m0 = blockIdx.y * 128;
    const int n0 = blockIdx.x * WN;

    __shared__ bf16 As[2 * 128 * LDS_STRIDE];
    __shared__ bf16 Bs[2 * WN  * LDS_STRIDE];

    v8f acc[NT];
    #pragma unroll
    for (int j = 0; j < NT; ++j)
        #pragma unroll
        for (int i = 0; i < 8; ++i) acc[j][i] = 0.0f;

    // A staging role: thread -> (row ra, 16-wide k-half kh)
    const int ra = tid >> 1;
    const int kh = (tid & 1) * 16;
    const float* Abase = A + (size_t)(m0 + ra) * lda + kh;
    // B staging role: thread -> (k-pair kb2, NCB-wide n-chunk nc)
    const int kb2 = tid >> 4;            // 0..15 (k rows 2*kb2, 2*kb2+1)
    const int nc  = (tid & 15) * NCB;
    const float* B0 = B + (size_t)(kb2 * 2) * ldb + n0 + nc;
    const float* B1 = B0 + ldb;

    // prologue: load K-step 0
    float2 fa[8], fb0[NCB / 2], fb1[NCB / 2];
    #pragma unroll
    for (int p = 0; p < 8; ++p) fa[p] = *(const float2*)(Abase + 2 * p);
    #pragma unroll
    for (int p = 0; p < NCB / 2; ++p) {
        fb0[p] = *(const float2*)(B0 + 2 * p);
        fb1[p] = *(const float2*)(B1 + 2 * p);
    }

    int buf = 0;
    for (int k = 0; k < K; k += 32) {
        bf16* Asb = As + buf * (128 * LDS_STRIDE);
        bf16* Bsb = Bs + buf * (WN  * LDS_STRIDE);

        // convert current tile + store to LDS
        uint4 pk0, pk1;
        pk0.x = pk2bf(fa[0].x, fa[0].y); pk0.y = pk2bf(fa[1].x, fa[1].y);
        pk0.z = pk2bf(fa[2].x, fa[2].y); pk0.w = pk2bf(fa[3].x, fa[3].y);
        pk1.x = pk2bf(fa[4].x, fa[4].y); pk1.y = pk2bf(fa[5].x, fa[5].y);
        pk1.z = pk2bf(fa[6].x, fa[6].y); pk1.w = pk2bf(fa[7].x, fa[7].y);
        *(uint4*)(Asb + ra * LDS_STRIDE + kh)     = pk0;   // 2x ds_store_b128
        *(uint4*)(Asb + ra * LDS_STRIDE + kh + 8) = pk1;
        #pragma unroll
        for (int c = 0; c < NCB; ++c) {   // transpose-on-store, k-pairs packed
            float x0 = (c & 1) ? fb0[c >> 1].y : fb0[c >> 1].x;
            float x1 = (c & 1) ? fb1[c >> 1].y : fb1[c >> 1].x;
            *(unsigned*)(Bsb + (nc + c) * LDS_STRIDE + kb2 * 2) = pk2bf(x0, x1);
        }

        __syncthreads();   // single barrier per K step (double buffered)

        // prefetch next K step's tiles; latency overlaps the WMMA phase
        if (k + 32 < K) {
            #pragma unroll
            for (int p = 0; p < 8; ++p)
                fa[p] = *(const float2*)(Abase + (k + 32) + 2 * p);
            #pragma unroll
            for (int p = 0; p < NCB / 2; ++p) {
                fb0[p] = *(const float2*)(B0 + (size_t)(k + 32) * ldb + 2 * p);
                fb1[p] = *(const float2*)(B1 + (size_t)(k + 32) * ldb + 2 * p);
            }
        }
        if (k + 64 < K) {  // warm GL2 two steps ahead (global_prefetch_b8)
            __builtin_prefetch(Abase + (k + 64), 0, 1);
            __builtin_prefetch(B0 + (size_t)(k + 64) * ldb, 0, 1);
        }

        // fragments + WMMA
        v16bf afr = ld_frag_a(Asb, wave * 16 + l15, half);
        #pragma unroll
        for (int j = 0; j < NT; ++j) {
            v16bf bfr = ld_frag_b(Bsb, j * 16 + l15, half);
            acc[j] = __builtin_amdgcn_wmma_f32_16x16x32_bf16(
                false, afr, false, bfr, (short)0, acc[j], false, false);
        }
        buf ^= 1;
    }

    #pragma unroll
    for (int j = 0; j < NT; ++j)
        #pragma unroll
        for (int i = 0; i < 8; ++i) {
            int row = m0 + wave * 16 + i + 8 * half;
            int col = n0 + j * 16 + l15;
            size_t idx = (size_t)row * ldd + col;
            float v = acc[j][i];
            if constexpr (MODE == EP_ADD) {
                v += aux[idx];
            } else if constexpr (MODE == EP_SILUMUL) {
                float g = aux[idx];
                v *= g / (1.0f + __expf(-g));
            } else if constexpr (MODE == EP_SCALEACC) {
                v = D[idx] + scalev[(size_t)row * sstride] * v;
            }
            D[idx] = v;
        }
}

// =======================================================================
// Flash-style causal MLA attention. One wave per (16-query tile, head).
// Q: [T, HEADS*192] (rope applied). KV: [T, HEADS*256] (k_nope|v).
// KPE: [T, 64] roped, broadcast across heads. O: [T, HEADS*128].
// =======================================================================
__global__ __launch_bounds__(32)
void mla_attention(const float* __restrict__ Q,
                   const float* __restrict__ KV,
                   const float* __restrict__ KPE,
                   float* __restrict__ O)
{
    const int h    = blockIdx.y;
    const int q0   = blockIdx.x * 16;
    const int lane = threadIdx.x;
    const int half = lane >> 4;
    const int l15  = lane & 15;
    const float sc = rsqrtf((float)QKD);

    __shared__ float pbuf[16 * 32];   // P transpose staging (C-layout -> A-layout)

    // Q fragments: 6 K-chunks of 32 covering d=192, resident in VGPRs
    v16bf qf[6];
    const float* Qrow = Q + (size_t)(q0 + l15) * (HEADS * QKD) + h * QKD;
    #pragma unroll
    for (int c = 0; c < 6; ++c)
        #pragma unroll
        for (int p = 0; p < 8; ++p) {
            int e = p * 2;
            int feat = c * 32 + half * 16 + e;
            float2 f = *(const float2*)(Qrow + feat);
            qf[c][e]     = f2bf(f.x);
            qf[c][e + 1] = f2bf(f.y);
        }

    v8f o[8];
    float mrun[8], lrun[8];
    #pragma unroll
    for (int j = 0; j < 8; ++j)
        #pragma unroll
        for (int i = 0; i < 8; ++i) o[j][i] = 0.0f;
    #pragma unroll
    for (int i = 0; i < 8; ++i) { mrun[i] = -1e30f; lrun[i] = 0.0f; }

    const int qmax = q0 + 15;
    for (int kb = 0; kb <= qmax; kb += 32) {
        // ---- S = Q K^T for 32 keys (two 16x16 tiles) ----
        v8f s[2];
        #pragma unroll
        for (int g = 0; g < 2; ++g)
            #pragma unroll
            for (int i = 0; i < 8; ++i) s[g][i] = 0.0f;

        #pragma unroll
        for (int c = 0; c < 6; ++c)
            #pragma unroll
            for (int g = 0; g < 2; ++g) {
                int key = kb + g * 16 + l15;
                if (key >= TSEQ) key = TSEQ - 1;       // clamped loads masked below
                int base = c * 32 + half * 16;
                const float* src = (c < 4)
                    ? KV  + (size_t)key * (HEADS * 256) + h * 256 + base
                    : KPE + (size_t)key * ROPE_D + (base - NOPE);
                v16bf b;
                #pragma unroll
                for (int p = 0; p < 8; ++p) {
                    float2 f = *(const float2*)(src + p * 2);
                    b[2 * p]     = f2bf(f.x);
                    b[2 * p + 1] = f2bf(f.y);
                }
                s[g] = __builtin_amdgcn_wmma_f32_16x16x32_bf16(
                    false, qf[c], false, b, (short)0, s[g], false, false);
            }

        // ---- scale + causal mask ----
        #pragma unroll
        for (int g = 0; g < 2; ++g)
            #pragma unroll
            for (int i = 0; i < 8; ++i) {
                int qrow = q0 + i + 8 * half;
                int key  = kb + g * 16 + l15;
                s[g][i] = (key <= qrow) ? s[g][i] * sc : -1e30f;
            }

        // ---- online softmax: row stats via 16-lane xor shuffles ----
        float alpha[8];
        #pragma unroll
        for (int i = 0; i < 8; ++i) {
            float mx = fmaxf(s[0][i], s[1][i]);
            #pragma unroll
            for (int d = 1; d < 16; d <<= 1) mx = fmaxf(mx, __shfl_xor(mx, d, 32));
            float mn = fmaxf(mrun[i], mx);
            alpha[i] = __expf(mrun[i] - mn);
            mrun[i]  = mn;
            float p0 = __expf(s[0][i] - mn);
            float p1 = __expf(s[1][i] - mn);
            s[0][i] = p0; s[1][i] = p1;
            float rs = p0 + p1;
            #pragma unroll
            for (int d = 1; d < 16; d <<= 1) rs += __shfl_xor(rs, d, 32);
            lrun[i] = lrun[i] * alpha[i] + rs;
        }

        // ---- transpose P through LDS (C-layout -> A-layout) ----
        #pragma unroll
        for (int g = 0; g < 2; ++g)
            #pragma unroll
            for (int i = 0; i < 8; ++i)
                pbuf[(i + 8 * half) * 32 + g * 16 + l15] = s[g][i];
        __syncthreads();
        v16bf pf;
        #pragma unroll
        for (int e = 0; e < 16; ++e) {
            int kk = half * 16 + e;
            pf[e] = f2bf(pbuf[l15 * 32 + kk]);
        }
        __syncthreads();

        // ---- O = diag(alpha) O + P V ----
        #pragma unroll
        for (int j = 0; j < 8; ++j) {
            #pragma unroll
            for (int i = 0; i < 8; ++i) o[j][i] *= alpha[i];
            const float* Vb = KV + h * 256 + NOPE + (j * 16 + l15);
            v16bf b;
            #pragma unroll
            for (int e = 0; e < 16; ++e) {
                int key = kb + half * 16 + e;
                if (key >= TSEQ) key = TSEQ - 1;       // p==0 for masked keys
                b[e] = f2bf(Vb[(size_t)key * (HEADS * 256)]);
            }
            o[j] = __builtin_amdgcn_wmma_f32_16x16x32_bf16(
                false, pf, false, b, (short)0, o[j], false, false);
        }
    }

    #pragma unroll
    for (int j = 0; j < 8; ++j)
        #pragma unroll
        for (int i = 0; i < 8; ++i) {
            int row = q0 + i + 8 * half;
            O[(size_t)row * (HEADS * VDIM) + h * VDIM + j * 16 + l15] = o[j][i] / lrun[i];
        }
}

// =======================================================================
// residual = hs + res ; h = rms(residual) * w     (one block per token)
// =======================================================================
__global__ __launch_bounds__(256)
void add_rms_kernel(const float* __restrict__ hs, const float* __restrict__ res,
                    const float* __restrict__ w,
                    float* __restrict__ r_out, float* __restrict__ h_out)
{
    const int t = blockIdx.x;
    __shared__ float red[8];
    float vals[HID / 256];
    float loc = 0.0f;
    #pragma unroll
    for (int i = 0; i < HID / 256; ++i) {
        int c = i * 256 + threadIdx.x;
        float v = hs[(size_t)t * HID + c] + res[(size_t)t * HID + c];
        r_out[(size_t)t * HID + c] = v;
        vals[i] = v;
        loc += v * v;
    }
    #pragma unroll
    for (int d = 1; d < 32; d <<= 1) loc += __shfl_xor(loc, d, 32);
    if ((threadIdx.x & 31) == 0) red[threadIdx.x >> 5] = loc;
    __syncthreads();
    float tot = 0.0f;
    #pragma unroll
    for (int i = 0; i < 8; ++i) tot += red[i];
    float rn = rsqrtf(tot / (float)HID + EPSV);
    #pragma unroll
    for (int i = 0; i < HID / 256; ++i) {
        int c = i * 256 + threadIdx.x;
        h_out[(size_t)t * HID + c] = vals[i] * rn * w[c];
    }
}

// generic RMSNorm over a row segment (supports src==dst in-place)
__global__ __launch_bounds__(256)
void rms_kernel(const float* __restrict__ src, int ld_src,
                float* __restrict__ dst, int ld_dst,
                const float* __restrict__ w, int width)
{
    const int t = blockIdx.x;
    const float* s = src + (size_t)t * ld_src;
    float* d = dst + (size_t)t * ld_dst;
    __shared__ float red[8];
    float loc = 0.0f;
    for (int c = threadIdx.x; c < width; c += 256) { float v = s[c]; loc += v * v; }
    #pragma unroll
    for (int dd = 1; dd < 32; dd <<= 1) loc += __shfl_xor(loc, dd, 32);
    if ((threadIdx.x & 31) == 0) red[threadIdx.x >> 5] = loc;
    __syncthreads();
    float tot = 0.0f;
    #pragma unroll
    for (int i = 0; i < 8; ++i) tot += red[i];
    float rn = rsqrtf(tot / (float)width + EPSV);
    for (int c = threadIdx.x; c < width; c += 256) d[c] = s[c] * rn * w[c];
}

// rope on k_pe slice of qkv -> KPE buffer
__global__ __launch_bounds__(32)
void rope_kpe_kernel(const int* __restrict__ pos, const float* __restrict__ qkv,
                     float* __restrict__ kpe)
{
    const int t = blockIdx.x, i = threadIdx.x;   // i in [0,32)
    const float* src = qkv + (size_t)t * QKV_N + Q_LORA + KV_LORA;
    float inv = __powf(ROPE_BASE, -(2.0f * (float)i) / (float)ROPE_D);
    float ang = (float)pos[t] * inv;
    float c = __cosf(ang), sn = __sinf(ang);
    float x1 = src[i], x2 = src[i + 32];
    kpe[(size_t)t * ROPE_D + i]      = x1 * c - x2 * sn;
    kpe[(size_t)t * ROPE_D + i + 32] = x2 * c + x1 * sn;
}

// in-place rope on q_pe slices of q buffer [T, HEADS*192]
__global__ __launch_bounds__(32)
void rope_q_kernel(const int* __restrict__ pos, float* __restrict__ q)
{
    const int t = blockIdx.x, h = blockIdx.y, i = threadIdx.x;
    float* p = q + (size_t)t * (HEADS * QKD) + h * QKD + NOPE;
    float inv = __powf(ROPE_BASE, -(2.0f * (float)i) / (float)ROPE_D);
    float ang = (float)pos[t] * inv;
    float c = __cosf(ang), sn = __sinf(ang);
    float x1 = p[i], x2 = p[i + 32];
    p[i]      = x1 * c - x2 * sn;
    p[i + 32] = x2 * c + x1 * sn;
}

// softmax(h @ gate_w) -> normalized top-2 combine weights (one wave/token)
__global__ __launch_bounds__(32)
void gate_top2_kernel(const float* __restrict__ h, const float* __restrict__ gw,
                      float* __restrict__ comb)
{
    const int t = blockIdx.x, lane = threadIdx.x;
    float acc[N_EXP];
    #pragma unroll
    for (int e = 0; e < N_EXP; ++e) acc[e] = 0.0f;
    for (int f = lane; f < HID; f += 32) {
        float hv = h[(size_t)t * HID + f];
        #pragma unroll
        for (int e = 0; e < N_EXP; ++e) acc[e] += hv * gw[(size_t)f * N_EXP + e];
    }
    #pragma unroll
    for (int e = 0; e < N_EXP; ++e)
        #pragma unroll
        for (int d = 1; d < 32; d <<= 1) acc[e] += __shfl_xor(acc[e], d, 32);
    if (lane == 0) {
        float mx = acc[0];
        #pragma unroll
        for (int e = 1; e < N_EXP; ++e) mx = fmaxf(mx, acc[e]);
        float ex[N_EXP]; float sum = 0.0f;
        #pragma unroll
        for (int e = 0; e < N_EXP; ++e) { ex[e] = __expf(acc[e] - mx); sum += ex[e]; }
        #pragma unroll
        for (int e = 0; e < N_EXP; ++e) ex[e] /= sum;
        int i1 = 0;
        #pragma unroll
        for (int e = 1; e < N_EXP; ++e) if (ex[e] > ex[i1]) i1 = e;
        int i2 = (i1 == 0) ? 1 : 0;
        #pragma unroll
        for (int e = 0; e < N_EXP; ++e) if (e != i1 && ex[e] > ex[i2]) i2 = e;
        float s2 = ex[i1] + ex[i2];
        #pragma unroll
        for (int e = 0; e < N_EXP; ++e)
            comb[(size_t)t * N_EXP + e] =
                (e == i1) ? ex[i1] / s2 : ((e == i2) ? ex[i2] / s2 : 0.0f);
    }
}

// =======================================================================
extern "C" void kernel_launch(void* const* d_in, const int* in_sizes, int n_in,
                              void* d_out, int out_size, void* d_ws, size_t ws_size,
                              hipStream_t stream)
{
    (void)in_sizes; (void)n_in; (void)out_size; (void)ws_size;

    const int*   positions = (const int*)  d_in[0];
    const float* hs        = (const float*)d_in[1];
    const float* res       = (const float*)d_in[2];
    const float* w_in_ln   = (const float*)d_in[3];
    const float* w_post_ln = (const float*)d_in[4];
    const float* w_q_ln    = (const float*)d_in[5];
    const float* w_kv_ln   = (const float*)d_in[6];
    const float* w_qkv_a   = (const float*)d_in[7];
    const float* w_q_b     = (const float*)d_in[8];
    const float* w_kv_b    = (const float*)d_in[9];
    const float* w_o       = (const float*)d_in[10];
    const float* gate_w    = (const float*)d_in[11];
    const float* we_gate   = (const float*)d_in[12];
    const float* we_up     = (const float*)d_in[13];
    const float* we_down   = (const float*)d_in[14];
    const float* ws_gate   = (const float*)d_in[15];
    const float* ws_up     = (const float*)d_in[16];
    const float* ws_down   = (const float*)d_in[17];

    // workspace layout (~95 MB of fp32)
    float* ws   = (float*)d_ws;
    float* RES1 = ws;                                  // [T, HID]
    float* H1   = RES1 + (size_t)TSEQ * HID;           // [T, HID]
    float* QKVB = H1   + (size_t)TSEQ * HID;           // [T, 2112]
    float* QB   = QKVB + (size_t)TSEQ * QKV_N;         // [T, HEADS*192]
    float* KVB  = QB   + (size_t)TSEQ * HEADS * QKD;   // [T, HEADS*256]
    float* KPE  = KVB  + (size_t)TSEQ * HEADS * (NOPE + VDIM); // [T, 64]
    float* ATTN = KPE  + (size_t)TSEQ * ROPE_D;        // [T, HID]
    float* H2   = ATTN + (size_t)TSEQ * HID;           // [T, HID]
    float* COMB = H2   + (size_t)TSEQ * HID;           // [T, 8]
    float* GBUF = COMB + (size_t)TSEQ * N_EXP;         // [T, 2816] (gate proj, reused)
    float* ABUF = GBUF + (size_t)TSEQ * SH_I;          // [T, 2816] (silu(g)*u, reused)

    float* OUT1   = (float*)d_out;                 // routed + shared  [T, HID]
    float* OUTRES = OUT1 + (size_t)TSEQ * HID;     // residual out     [T, HID]

    const dim3 blk(256);
    auto G = [](int N) { return dim3(N / 128, TSEQ / 128); };

    // 1) residual + input norm
    add_rms_kernel<<<TSEQ, 256, 0, stream>>>(hs, res, w_in_ln, RES1, H1);
    // 2) qkv_a projection: 2048-wide main + 64-wide tail (2112 = 16*128 + 64)
    wmma_gemm<128, EP_STORE><<<G(2048), blk, 0, stream>>>(
        H1, HID, w_qkv_a, QKV_N, QKVB, QKV_N, HID, nullptr, nullptr, 0);
    wmma_gemm<64, EP_STORE><<<dim3(1, TSEQ / 128), blk, 0, stream>>>(
        H1, HID, w_qkv_a + 2048, QKV_N, QKVB + 2048, QKV_N, HID, nullptr, nullptr, 0);
    // 3) per-part norms + rope(k_pe)
    rms_kernel<<<TSEQ, 256, 0, stream>>>(QKVB, QKV_N, QKVB, QKV_N, w_q_ln, Q_LORA);
    rms_kernel<<<TSEQ, 256, 0, stream>>>(QKVB + Q_LORA, QKV_N, QKVB + Q_LORA, QKV_N,
                                         w_kv_ln, KV_LORA);
    rope_kpe_kernel<<<TSEQ, 32, 0, stream>>>(positions, QKVB, KPE);
    // 4) q / kv up-projections
    wmma_gemm<128, EP_STORE><<<G(HEADS * QKD), blk, 0, stream>>>(
        QKVB, QKV_N, w_q_b, HEADS * QKD, QB, HEADS * QKD, Q_LORA, nullptr, nullptr, 0);
    wmma_gemm<128, EP_STORE><<<G(HEADS * 256), blk, 0, stream>>>(
        QKVB + Q_LORA, QKV_N, w_kv_b, HEADS * 256, KVB, HEADS * 256, KV_LORA,
        nullptr, nullptr, 0);
    rope_q_kernel<<<dim3(TSEQ, HEADS), 32, 0, stream>>>(positions, QB);
    // 5) flash MLA attention
    mla_attention<<<dim3(TSEQ / 16, HEADS), 32, 0, stream>>>(QB, KVB, KPE, ATTN);
    // 6) output projection + residual (writes residual output directly)
    wmma_gemm<128, EP_ADD><<<G(HID), blk, 0, stream>>>(
        ATTN, HID, w_o, HID, OUTRES, HID, HID, RES1, nullptr, 0);
    // 7) post norm + router
    rms_kernel<<<TSEQ, 256, 0, stream>>>(OUTRES, HID, H2, HID, w_post_ln, HID);
    gate_top2_kernel<<<TSEQ, 32, 0, stream>>>(H2, gate_w, COMB);
    // 8) shared expert -> OUT1
    wmma_gemm<128, EP_STORE><<<G(SH_I), blk, 0, stream>>>(
        H2, HID, ws_gate, SH_I, GBUF, SH_I, HID, nullptr, nullptr, 0);
    wmma_gemm<128, EP_SILUMUL><<<G(SH_I), blk, 0, stream>>>(
        H2, HID, ws_up, SH_I, ABUF, SH_I, HID, GBUF, nullptr, 0);
    wmma_gemm<128, EP_STORE><<<G(HID), blk, 0, stream>>>(
        ABUF, SH_I, ws_down, HID, OUT1, HID, SH_I, nullptr, nullptr, 0);
    // 9) routed experts, comb-weighted accumulate into OUT1
    for (int e = 0; e < N_EXP; ++e) {
        const float* weg = we_gate + (size_t)e * HID * MOE_I;
        const float* weu = we_up   + (size_t)e * HID * MOE_I;
        const float* wed = we_down + (size_t)e * MOE_I * HID;
        wmma_gemm<128, EP_STORE><<<G(MOE_I), blk, 0, stream>>>(
            H2, HID, weg, MOE_I, GBUF, MOE_I, HID, nullptr, nullptr, 0);
        wmma_gemm<128, EP_SILUMUL><<<G(MOE_I), blk, 0, stream>>>(
            H2, HID, weu, MOE_I, ABUF, MOE_I, HID, GBUF, nullptr, 0);
        wmma_gemm<128, EP_SCALEACC><<<G(HID), blk, 0, stream>>>(
            ABUF, MOE_I, wed, HID, OUT1, HID, MOE_I, nullptr, COMB + e, N_EXP);
    }
}
